// PhysicsGPSLayer_39728447488412
// MI455X (gfx1250) — matrix operand order repaired and checked by reading, not compiled, not run on hardware
//
#include <hip/hip_runtime.h>
#include <hip/hip_bf16.h>
#include <math.h>

// Problem constants (fixed by the reference).
#define NN  16384
#define DD  128
#define EDD 32
#define EE  262144
#define GG  32
#define SS  512
#define HH  4
#define HDIM 32

typedef __attribute__((ext_vector_type(16))) _Float16 v16h;
typedef __attribute__((ext_vector_type(8)))  _Float16 v8h;
typedef __attribute__((ext_vector_type(8)))  float    v8f;

// ---------------------------------------------------------------- WMMA core
static __device__ __forceinline__ v8f wmma16(v16h a, v16h b, v8f c) {
  // D = A(16x32 f16) * B(32x16 f16) + C(16x16 f32)
  return __builtin_amdgcn_wmma_f32_16x16x32_f16(false, a, false, b, (short)0, c,
                                                false, false);
}
static __device__ __forceinline__ v16h cat8(v8h lo, v8h hi) {
  return __builtin_shufflevector(lo, hi, 0, 1, 2, 3, 4, 5, 6, 7,
                                 8, 9, 10, 11, 12, 13, 14, 15);
}
static __device__ __forceinline__ v8f vzero8() {
  v8f v;
#pragma unroll
  for (int i = 0; i < 8; ++i) v[i] = 0.f;
  return v;
}
static __device__ __forceinline__ v8f vsplat8(float x) {
  v8f v;
#pragma unroll
  for (int i = 0; i < 8; ++i) v[i] = x;
  return v;
}

// ------------------------------------------------------- fragment loaders
// A-frag (16x32 f16): lane m=lane&15 is row; K map: j -> 16g+j
static __device__ __forceinline__ v16h load_a_rowp_f16(const _Float16* rowp, int kbase) {
  const int g = (threadIdx.x & 31) >> 4;
  const _Float16* r = rowp + kbase + 16 * g;
  v8h lo = *(const v8h*)(r);
  v8h hi = *(const v8h*)(r + 8);
  return cat8(lo, hi);
}
static __device__ __forceinline__ v16h load_a_rowp_f32(const float* rowp, int kbase) {
  const int g = (threadIdx.x & 31) >> 4;
  const float* r = rowp + kbase + 16 * g;
  v16h a;
#pragma unroll
  for (int j = 0; j < 8; ++j) a[j] = (_Float16)r[j];
#pragma unroll
  for (int j = 8; j < 16; ++j) a[j] = (_Float16)r[j];
  return a;
}
// B-frag from PRE-SWIZZLED weight: frag f=(c*ntn+nt) stored as [f][lane][16 halves]
static __device__ __forceinline__ v16h load_b_swz(const _Float16* wz, int ntn, int c, int nt) {
  const int lane = threadIdx.x & 31;
  const _Float16* p = wz + ((((size_t)c * ntn + nt) * 32 + lane) << 4);
  v8h lo = *(const v8h*)(p);
  v8h hi = *(const v8h*)(p + 8);
  return cat8(lo, hi);
}
// B-frag of K^T from LDS key-major slab [key][32]: contiguous per lane.
static __device__ __forceinline__ v16h load_b_kt(const _Float16* slab, int key0) {
  const int lane = threadIdx.x & 31;
  const int n = lane & 15, g = lane >> 4;
  const _Float16* p = slab + (size_t)(key0 + n) * HDIM + 16 * g;
  v8h lo = *(const v8h*)(p);
  v8h hi = *(const v8h*)(p + 8);
  return cat8(lo, hi);
}
// B-frag of V from LDS PRE-SWIZZLED fragment buffer: frag f=(kc*2+nt).
static __device__ __forceinline__ v16h load_b_vz(const _Float16* vz, int kc, int nt) {
  const int lane = threadIdx.x & 31;
  const _Float16* p = vz + (((size_t)(kc * 2 + nt) * 32 + lane) << 4);
  v8h lo = *(const v8h*)(p);
  v8h hi = *(const v8h*)(p + 8);
  return cat8(lo, hi);
}

// ---------------------------------------------------------------- helpers
// Branchless erf (Abramowitz-Stegun 7.1.26, |err| <= 1.5e-7): rcp + exp + 5 fma.
static __device__ __forceinline__ float erf_fast(float x) {
  const float ax = fabsf(x);
  const float t = __builtin_amdgcn_rcpf(1.f + 0.3275911f * ax);
  float y = 1.061405429f;
  y = y * t - 1.453152027f;
  y = y * t + 1.421413741f;
  y = y * t - 0.284496736f;
  y = y * t + 0.254829592f;
  const float r = 1.f - (y * t) * __expf(-ax * ax);
  return copysignf(r, x);
}
static __device__ __forceinline__ float gelu_f(float x) {
  return 0.5f * x * (1.f + erf_fast(x * 0.70710678118654752f));
}
static __device__ __forceinline__ float rowmax16(float v) {
  v = fmaxf(v, __shfl_xor(v, 8, 32));
  v = fmaxf(v, __shfl_xor(v, 4, 32));
  v = fmaxf(v, __shfl_xor(v, 2, 32));
  v = fmaxf(v, __shfl_xor(v, 1, 32));
  return v;
}
static __device__ __forceinline__ float rowsum16(float v) {
  v += __shfl_xor(v, 8, 32);
  v += __shfl_xor(v, 4, 32);
  v += __shfl_xor(v, 2, 32);
  v += __shfl_xor(v, 1, 32);
  return v;
}
static __device__ __forceinline__ float wavesum32(float v) {
  v += __shfl_xor(v, 16, 32);
  v += __shfl_xor(v, 8, 32);
  v += __shfl_xor(v, 4, 32);
  v += __shfl_xor(v, 2, 32);
  v += __shfl_xor(v, 1, 32);
  return v;
}
// 16B global->LDS async copy (GLOBAL_LOAD_ASYNC_TO_LDS_B128, GV mode), ASYNCcnt-tracked.
static __device__ __forceinline__ void async_copy16(void* lds_dst, const void* gsrc) {
  unsigned ldsoff = (unsigned)(unsigned long long)lds_dst;  // LDS aperture: low 32 = offset
  asm volatile("global_load_async_to_lds_b128 %0, %1, off"
               :: "v"(ldsoff), "v"(gsrc) : "memory");
}
static __device__ __forceinline__ void wait_async0() {
#if __has_builtin(__builtin_amdgcn_s_wait_asynccnt)
  __builtin_amdgcn_s_wait_asynccnt(0);
#else
  asm volatile("s_wait_asynccnt 0" ::: "memory");
#endif
}

// ---------------------------------------------------------------- kernels
__global__ __launch_bounds__(256) void cvt_f16_kernel(const float* __restrict__ in,
                                                      _Float16* __restrict__ out, int n) {
  int i = blockIdx.x * blockDim.x + threadIdx.x;
  if (i < n) out[i] = (_Float16)in[i];
}

// Swizzle a KxNc row-major f32 weight into f16 B-fragment order.
__global__ __launch_bounds__(256) void swz_w_kernel(const float* __restrict__ w,
                                                    _Float16* __restrict__ out,
                                                    int K, int Nc) {
  int idx = blockIdx.x * blockDim.x + threadIdx.x;
  if (idx >= K * Nc) return;
  const int j = idx & 15;
  const int L = (idx >> 4) & 31;
  const int f = idx >> 9;
  const int ntn = Nc >> 4;
  const int nt = f % ntn, c = f / ntn;
  const int n = L & 15, g = L >> 4;
  const int k = 32 * c + 16 * g + j;
  out[idx] = (_Float16)w[(size_t)k * Nc + nt * 16 + n];
}

// LayerNorm over D=128; one wave per row; optional f32 / f16 outputs
__global__ __launch_bounds__(256) void ln_kernel(const float* __restrict__ in,
                                                 const float* __restrict__ gamma,
                                                 const float* __restrict__ beta,
                                                 float* __restrict__ out32,
                                                 _Float16* __restrict__ out16) {
  const int lane = threadIdx.x & 31;
  const int row = blockIdx.x * 8 + (threadIdx.x >> 5);
  const float* p = in + (size_t)row * DD;
  float v[4];
#pragma unroll
  for (int i = 0; i < 4; ++i) v[i] = p[lane + 32 * i];
  float mean = wavesum32(v[0] + v[1] + v[2] + v[3]) * (1.f / DD);
  float d2 = 0.f;
#pragma unroll
  for (int i = 0; i < 4; ++i) { float d = v[i] - mean; d2 += d * d; }
  float rstd = rsqrtf(wavesum32(d2) * (1.f / DD) + 1e-5f);
#pragma unroll
  for (int i = 0; i < 4; ++i) {
    int c = lane + 32 * i;
    float o = (v[i] - mean) * rstd * gamma[c] + beta[c];
    if (out32) out32[(size_t)row * DD + c] = o;
    if (out16) out16[(size_t)row * DD + c] = (_Float16)o;
  }
}

// Message MLP over edges + scatter-add: 16 edges per wave, 4 waves per block.
// N-tiles processed in pairs (two accumulators) to overlap B loads with WMMA.
__global__ __launch_bounds__(128) void msg_kernel(const _Float16* __restrict__ xn16,
                                                  const int* __restrict__ ei,
                                                  const float* __restrict__ ea,
                                                  const _Float16* __restrict__ W1z,
                                                  const float* __restrict__ b1,
                                                  const _Float16* __restrict__ W2z,
                                                  const float* __restrict__ b2,
                                                  float* __restrict__ aggr) {
  __shared__ __align__(32) _Float16 Hs[4][16 * 256];
  const int lane = threadIdx.x & 31, wv = threadIdx.x >> 5;
  const int n = lane & 15, half = lane >> 4;
  const int e0 = (blockIdx.x * 4 + wv) * 16;

  const int srcRow = ei[e0 + n];
  const int dstRow = ei[EE + e0 + n];
  const _Float16* pd = xn16 + (size_t)dstRow * DD;
  const _Float16* ps = xn16 + (size_t)srcRow * DD;
  const float* pe = ea + (size_t)(e0 + n) * EDD;

  v16h A[9];
#pragma unroll
  for (int c = 0; c < 4; ++c) A[c] = load_a_rowp_f16(pd, 32 * c);
#pragma unroll
  for (int c = 0; c < 4; ++c) A[4 + c] = load_a_rowp_f16(ps, 32 * c);
  A[8] = load_a_rowp_f32(pe, 0);

  _Float16* hs = Hs[wv];
  for (int nt = 0; nt < 16; nt += 2) {              // hidden = 256
    v8f acc0 = vsplat8(b1[nt * 16 + n]);
    v8f acc1 = vsplat8(b1[(nt + 1) * 16 + n]);
#pragma unroll
    for (int c = 0; c < 9; ++c) {
      v16h B0 = load_b_swz(W1z, 16, c, nt);
      v16h B1 = load_b_swz(W1z, 16, c, nt + 1);
      acc0 = wmma16(A[c], B0, acc0);
      acc1 = wmma16(A[c], B1, acc1);
    }
#pragma unroll
    for (int r = 0; r < 8; ++r) {
      hs[(r + 8 * half) * 256 + nt * 16 + n]        = (_Float16)gelu_f(acc0[r]);
      hs[(r + 8 * half) * 256 + (nt + 1) * 16 + n]  = (_Float16)gelu_f(acc1[r]);
    }
  }

  v16h A2[8];
#pragma unroll
  for (int c = 0; c < 8; ++c) A2[c] = load_a_rowp_f16(hs + n * 256, 32 * c);

  int drow[8];
#pragma unroll
  for (int r = 0; r < 8; ++r) drow[r] = ei[EE + e0 + 8 * half + r];

  for (int nt = 0; nt < 8; nt += 2) {               // out = 128
    v8f acc0 = vsplat8(b2[nt * 16 + n]);
    v8f acc1 = vsplat8(b2[(nt + 1) * 16 + n]);
#pragma unroll
    for (int c = 0; c < 8; ++c) {
      v16h B0 = load_b_swz(W2z, 8, c, nt);
      v16h B1 = load_b_swz(W2z, 8, c, nt + 1);
      acc0 = wmma16(A2[c], B0, acc0);
      acc1 = wmma16(A2[c], B1, acc1);
    }
#pragma unroll
    for (int r = 0; r < 8; ++r) {
      atomicAdd(&aggr[(size_t)drow[r] * DD + nt * 16 + n], acc0[r]);
      atomicAdd(&aggr[(size_t)drow[r] * DD + (nt + 1) * 16 + n], acc1[r]);
    }
  }
}

// Update MLP: h_local = gelu([xn, aggr] @ U1 + b1) @ U2 + b2
__global__ __launch_bounds__(128) void upd_kernel(const _Float16* __restrict__ xn16,
                                                  const float* __restrict__ aggr,
                                                  const _Float16* __restrict__ U1z,
                                                  const float* __restrict__ b1,
                                                  const _Float16* __restrict__ U2z,
                                                  const float* __restrict__ b2,
                                                  float* __restrict__ hloc) {
  __shared__ __align__(32) _Float16 Hs[4][16 * 128];
  const int lane = threadIdx.x & 31, wv = threadIdx.x >> 5;
  const int n = lane & 15, half = lane >> 4;
  const int row0 = (blockIdx.x * 4 + wv) * 16;

  const _Float16* px = xn16 + (size_t)(row0 + n) * DD;
  const float* pa = aggr + (size_t)(row0 + n) * DD;
  v16h A[8];
#pragma unroll
  for (int c = 0; c < 4; ++c) A[c] = load_a_rowp_f16(px, 32 * c);
#pragma unroll
  for (int c = 0; c < 4; ++c) A[4 + c] = load_a_rowp_f32(pa, 32 * c);

  _Float16* hs = Hs[wv];
  for (int nt = 0; nt < 8; nt += 2) {
    v8f acc0 = vsplat8(b1[nt * 16 + n]);
    v8f acc1 = vsplat8(b1[(nt + 1) * 16 + n]);
#pragma unroll
    for (int c = 0; c < 8; ++c) {
      v16h B0 = load_b_swz(U1z, 8, c, nt);
      v16h B1 = load_b_swz(U1z, 8, c, nt + 1);
      acc0 = wmma16(A[c], B0, acc0);
      acc1 = wmma16(A[c], B1, acc1);
    }
#pragma unroll
    for (int r = 0; r < 8; ++r) {
      hs[(r + 8 * half) * 128 + nt * 16 + n]       = (_Float16)gelu_f(acc0[r]);
      hs[(r + 8 * half) * 128 + (nt + 1) * 16 + n] = (_Float16)gelu_f(acc1[r]);
    }
  }
  v16h A2[4];
#pragma unroll
  for (int c = 0; c < 4; ++c) A2[c] = load_a_rowp_f16(hs + n * 128, 32 * c);
  for (int nt = 0; nt < 8; nt += 2) {
    v8f acc0 = vsplat8(b2[nt * 16 + n]);
    v8f acc1 = vsplat8(b2[(nt + 1) * 16 + n]);
#pragma unroll
    for (int c = 0; c < 4; ++c) {
      v16h B0 = load_b_swz(U2z, 8, c, nt);
      v16h B1 = load_b_swz(U2z, 8, c, nt + 1);
      acc0 = wmma16(A2[c], B0, acc0);
      acc1 = wmma16(A2[c], B1, acc1);
    }
#pragma unroll
    for (int r = 0; r < 8; ++r) {
      hloc[(size_t)(row0 + r + 8 * half) * DD + nt * 16 + n] = acc0[r];
      hloc[(size_t)(row0 + r + 8 * half) * DD + (nt + 1) * 16 + n] = acc1[r];
    }
  }
}

// QKV projection; q pre-scaled by 1/sqrt(HD)
__global__ __launch_bounds__(128) void qkv_kernel(const _Float16* __restrict__ x16,
                                                  const _Float16* __restrict__ INz,
                                                  const float* __restrict__ inb,
                                                  _Float16* __restrict__ q16,
                                                  _Float16* __restrict__ k16,
                                                  _Float16* __restrict__ v16o) {
  const int lane = threadIdx.x & 31, wv = threadIdx.x >> 5;
  const int n = lane & 15, half = lane >> 4;
  const int row0 = (blockIdx.x * 4 + wv) * 16;
  const _Float16* px = x16 + (size_t)(row0 + n) * DD;
  v16h A[4];
#pragma unroll
  for (int c = 0; c < 4; ++c) A[c] = load_a_rowp_f16(px, 32 * c);

  for (int nt = 0; nt < 24; nt += 2) {              // 3D = 384 cols; pairs never straddle q/k/v
    v8f acc0 = vsplat8(inb[nt * 16 + n]);
    v8f acc1 = vsplat8(inb[(nt + 1) * 16 + n]);
#pragma unroll
    for (int c = 0; c < 4; ++c) {
      v16h B0 = load_b_swz(INz, 24, c, nt);
      v16h B1 = load_b_swz(INz, 24, c, nt + 1);
      acc0 = wmma16(A[c], B0, acc0);
      acc1 = wmma16(A[c], B1, acc1);
    }
#pragma unroll
    for (int u = 0; u < 2; ++u) {
      const int ntu = nt + u;
      const int c0 = ntu * 16;
#pragma unroll
      for (int r = 0; r < 8; ++r) {
        const size_t row = (size_t)(row0 + r + 8 * half);
        const float vv = (u == 0) ? acc0[r] : acc1[r];
        if (ntu < 8)       q16[row * DD + c0 + n] = (_Float16)(vv * 0.17677669529663689f);
        else if (ntu < 16) k16[row * DD + (c0 - 128) + n] = (_Float16)vv;
        else               v16o[row * DD + (c0 - 256) + n] = (_Float16)vv;
      }
    }
  }
}

// Flash attention: one block per (graph, head); K,V resident in LDS.
// K slab filled with async b128 copies (ASYNCcnt); V pre-swizzled to B-frag order.
__global__ __launch_bounds__(128) void attn_kernel(const _Float16* __restrict__ q16,
                                                   const _Float16* __restrict__ k16,
                                                   const _Float16* __restrict__ v16i,
                                                   _Float16* __restrict__ ao) {
  __shared__ __align__(32) _Float16 Ks[SS * HDIM];   // row-major [key][32]
  __shared__ __align__(32) _Float16 Vz[SS * HDIM];   // B-frag order [kc*2+nt][lane][16]
  __shared__ __align__(32) _Float16 Ps[4][16 * 32];
  const int g = blockIdx.x >> 2, h = blockIdx.x & 3;
  const int tid = threadIdx.x;
  const size_t gbase = (size_t)g * SS * DD + h * HDIM;

  for (int i = tid; i < SS * 4; i += 128) {
    const int key = i >> 2, part = i & 3;
    async_copy16(Ks + key * HDIM + part * 8,
                 k16 + gbase + (size_t)key * DD + part * 8);
    v8h vv = *(const v8h*)(v16i + gbase + (size_t)key * DD + part * 8);
    const int kc = key >> 5, gg = (key >> 4) & 1, j = key & 15;
#pragma unroll
    for (int e = 0; e < 8; ++e) {
      const int col = part * 8 + e;
      const int nt = col >> 4, nn = col & 15;
      Vz[(((kc * 2 + nt) * 32 + 16 * gg + nn) << 4) + j] = vv[e];
    }
  }
  wait_async0();
  __syncthreads();

  const int lane = tid & 31, wv = tid >> 5;
  const int n = lane & 15, half = lane >> 4;
  _Float16* ps = Ps[wv];

  for (int t = 0; t < 8; ++t) {                     // 128 queries per wave
    const int q0 = wv * 128 + t * 16;
    v16h qf = load_a_rowp_f16(q16 + gbase + (size_t)(q0 + n) * DD, 0);
    v8f o0 = vzero8(), o1 = vzero8();
    float rm[8], rs[8];
#pragma unroll
    for (int r = 0; r < 8; ++r) { rm[r] = -1e30f; rs[r] = 0.f; }

    for (int kc = 0; kc < 16; ++kc) {               // 32 keys per step
      const int key0 = kc * 32;
      v8f s0 = wmma16(qf, load_b_kt(Ks, key0), vzero8());
      v8f s1 = wmma16(qf, load_b_kt(Ks, key0 + 16), vzero8());
      float p0[8], p1[8];
#pragma unroll
      for (int r = 0; r < 8; ++r) {
        float cm = rowmax16(fmaxf(s0[r], s1[r]));
        float nm = fmaxf(rm[r], cm);
        float alpha = __expf(rm[r] - nm);
        float e0 = __expf(s0[r] - nm), e1 = __expf(s1[r] - nm);
        rs[r] = rs[r] * alpha + rowsum16(e0 + e1);
        rm[r] = nm;
        o0[r] *= alpha; o1[r] *= alpha;
        p0[r] = e0; p1[r] = e1;
      }
#pragma unroll
      for (int r = 0; r < 8; ++r) {                 // D-frag -> A-frag via per-wave LDS
        ps[(r + 8 * half) * 32 + n]      = (_Float16)p0[r];
        ps[(r + 8 * half) * 32 + 16 + n] = (_Float16)p1[r];
      }
      v16h pf = load_a_rowp_f16(ps + n * 32, 0);
      o0 = wmma16(pf, load_b_vz(Vz, kc, 0), o0);
      o1 = wmma16(pf, load_b_vz(Vz, kc, 1), o1);
    }
#pragma unroll
    for (int r = 0; r < 8; ++r) {
      const float inv = 1.f / rs[r];
      const size_t row = (size_t)(q0 + r + 8 * half);
      ao[gbase + row * DD + n]      = (_Float16)(o0[r] * inv);
      ao[gbase + row * DD + 16 + n] = (_Float16)(o1[r] * inv);
    }
  }
}

// out-proj + residual: t = x + h_local + (ao @ out_w + out_b)
__global__ __launch_bounds__(128) void outproj_kernel(const _Float16* __restrict__ ao,
                                                      const _Float16* __restrict__ OWz,
                                                      const float* __restrict__ ob,
                                                      const float* __restrict__ x,
                                                      const float* __restrict__ hloc,
                                                      float* __restrict__ t) {
  const int lane = threadIdx.x & 31, wv = threadIdx.x >> 5;
  const int n = lane & 15, half = lane >> 4;
  const int row0 = (blockIdx.x * 4 + wv) * 16;
  const _Float16* pa = ao + (size_t)(row0 + n) * DD;
  v16h A[4];
#pragma unroll
  for (int c = 0; c < 4; ++c) A[c] = load_a_rowp_f16(pa, 32 * c);
  for (int nt = 0; nt < 8; nt += 2) {
    v8f acc0 = vsplat8(ob[nt * 16 + n]);
    v8f acc1 = vsplat8(ob[(nt + 1) * 16 + n]);
#pragma unroll
    for (int c = 0; c < 4; ++c) {
      v16h B0 = load_b_swz(OWz, 8, c, nt);
      v16h B1 = load_b_swz(OWz, 8, c, nt + 1);
      acc0 = wmma16(A[c], B0, acc0);
      acc1 = wmma16(A[c], B1, acc1);
    }
#pragma unroll
    for (int r = 0; r < 8; ++r) {
      const size_t i0 = (size_t)(row0 + r + 8 * half) * DD + nt * 16 + n;
      const size_t i1 = i0 + 16;
      t[i0] = x[i0] + hloc[i0] + acc0[r];
      t[i1] = x[i1] + hloc[i1] + acc1[r];
    }
  }
}

// FFN: s = h1 + (gelu(h1 @ F1 + b1) @ F2 + b2)
__global__ __launch_bounds__(128) void ffn_kernel(const _Float16* __restrict__ h16,
                                                  const float* __restrict__ h32,
                                                  const _Float16* __restrict__ F1z,
                                                  const float* __restrict__ fb1,
                                                  const _Float16* __restrict__ F2z,
                                                  const float* __restrict__ fb2,
                                                  float* __restrict__ s) {
  __shared__ __align__(32) _Float16 Hs[4][16 * 256];
  const int lane = threadIdx.x & 31, wv = threadIdx.x >> 5;
  const int n = lane & 15, half = lane >> 4;
  const int row0 = (blockIdx.x * 4 + wv) * 16;
  const _Float16* px = h16 + (size_t)(row0 + n) * DD;
  v16h A[4];
#pragma unroll
  for (int c = 0; c < 4; ++c) A[c] = load_a_rowp_f16(px, 32 * c);

  _Float16* hs = Hs[wv];
  for (int nt = 0; nt < 16; nt += 2) {
    v8f acc0 = vsplat8(fb1[nt * 16 + n]);
    v8f acc1 = vsplat8(fb1[(nt + 1) * 16 + n]);
#pragma unroll
    for (int c = 0; c < 4; ++c) {
      v16h B0 = load_b_swz(F1z, 16, c, nt);
      v16h B1 = load_b_swz(F1z, 16, c, nt + 1);
      acc0 = wmma16(A[c], B0, acc0);
      acc1 = wmma16(A[c], B1, acc1);
    }
#pragma unroll
    for (int r = 0; r < 8; ++r) {
      hs[(r + 8 * half) * 256 + nt * 16 + n]       = (_Float16)gelu_f(acc0[r]);
      hs[(r + 8 * half) * 256 + (nt + 1) * 16 + n] = (_Float16)gelu_f(acc1[r]);
    }
  }
  v16h A2[8];
#pragma unroll
  for (int c = 0; c < 8; ++c) A2[c] = load_a_rowp_f16(hs + n * 256, 32 * c);
  for (int nt = 0; nt < 8; nt += 2) {
    v8f acc0 = vsplat8(fb2[nt * 16 + n]);
    v8f acc1 = vsplat8(fb2[(nt + 1) * 16 + n]);
#pragma unroll
    for (int c = 0; c < 8; ++c) {
      v16h B0 = load_b_swz(F2z, 8, c, nt);
      v16h B1 = load_b_swz(F2z, 8, c, nt + 1);
      acc0 = wmma16(A2[c], B0, acc0);
      acc1 = wmma16(A2[c], B1, acc1);
    }
#pragma unroll
    for (int r = 0; r < 8; ++r) {
      const size_t i0 = (size_t)(row0 + r + 8 * half) * DD + nt * 16 + n;
      const size_t i1 = i0 + 16;
      s[i0] = h32[i0] + acc0[r];
      s[i1] = h32[i1] + acc1[r];
    }
  }
}

// ---------------------------------------------------------------- launch
extern "C" void kernel_launch(void* const* d_in, const int* in_sizes, int n_in,
                              void* d_out, int out_size, void* d_ws, size_t ws_size,
                              hipStream_t stream) {
  const float* x      = (const float*)d_in[0];
  const int*   ei     = (const int*)d_in[1];
  const float* ea     = (const float*)d_in[2];
  const float* gnn_g  = (const float*)d_in[4];
  const float* gnn_b  = (const float*)d_in[5];
  const float* msg_w1 = (const float*)d_in[6];
  const float* msg_b1 = (const float*)d_in[7];
  const float* msg_w2 = (const float*)d_in[8];
  const float* msg_b2 = (const float*)d_in[9];
  const float* upd_w1 = (const float*)d_in[10];
  const float* upd_b1 = (const float*)d_in[11];
  const float* upd_w2 = (const float*)d_in[12];
  const float* upd_b2 = (const float*)d_in[13];
  const float* in_w   = (const float*)d_in[14];
  const float* in_b   = (const float*)d_in[15];
  const float* out_w  = (const float*)d_in[16];
  const float* out_b  = (const float*)d_in[17];
  const float* ffn_w1 = (const float*)d_in[18];
  const float* ffn_b1 = (const float*)d_in[19];
  const float* ffn_w2 = (const float*)d_in[20];
  const float* ffn_b2 = (const float*)d_in[21];
  const float* n1_g   = (const float*)d_in[22];
  const float* n1_b   = (const float*)d_in[23];
  const float* n2_g   = (const float*)d_in[24];
  const float* n2_b   = (const float*)d_in[25];

  char* ws = (char*)d_ws;
  size_t off = 0;
  auto alloc = [&](size_t bytes) -> void* {
    void* p = (void*)(ws + off);
    off += (bytes + 255) & ~(size_t)255;
    return p;
  };
  const size_t ND = (size_t)NN * DD;
  _Float16* W1z  = (_Float16*)alloc(288 * 256 * sizeof(_Float16));
  _Float16* W2z  = (_Float16*)alloc(256 * 128 * sizeof(_Float16));
  _Float16* U1z  = (_Float16*)alloc(256 * 128 * sizeof(_Float16));
  _Float16* U2z  = (_Float16*)alloc(128 * 128 * sizeof(_Float16));
  _Float16* INz  = (_Float16*)alloc(128 * 384 * sizeof(_Float16));
  _Float16* OWz  = (_Float16*)alloc(128 * 128 * sizeof(_Float16));
  _Float16* F1z  = (_Float16*)alloc(128 * 256 * sizeof(_Float16));
  _Float16* F2z  = (_Float16*)alloc(256 * 128 * sizeof(_Float16));
  _Float16* x16  = (_Float16*)alloc(ND * sizeof(_Float16));
  _Float16* xn16 = (_Float16*)alloc(ND * sizeof(_Float16));
  _Float16* q16  = (_Float16*)alloc(ND * sizeof(_Float16));
  _Float16* k16  = (_Float16*)alloc(ND * sizeof(_Float16));
  _Float16* v16b = (_Float16*)alloc(ND * sizeof(_Float16));
  _Float16* ao16 = (_Float16*)alloc(ND * sizeof(_Float16));
  _Float16* h116 = (_Float16*)alloc(ND * sizeof(_Float16));
  float* aggr  = (float*)alloc(ND * sizeof(float));
  float* hloc  = (float*)alloc(ND * sizeof(float));
  float* t32   = (float*)alloc(ND * sizeof(float));
  float* h132  = (float*)alloc(ND * sizeof(float));

  auto swz = [&](const float* src, _Float16* dst, int K, int Nc) {
    swz_w_kernel<<<(K * Nc + 255) / 256, 256, 0, stream>>>(src, dst, K, Nc);
  };
  swz(msg_w1, W1z, 288, 256);
  swz(msg_w2, W2z, 256, 128);
  swz(upd_w1, U1z, 256, 128);
  swz(upd_w2, U2z, 128, 128);
  swz(in_w,   INz, 128, 384);
  swz(out_w,  OWz, 128, 128);
  swz(ffn_w1, F1z, 128, 256);
  swz(ffn_w2, F2z, 256, 128);
  cvt_f16_kernel<<<(int)(ND + 255) / 256, 256, 0, stream>>>(x, x16, (int)ND);

  hipMemsetAsync(aggr, 0, ND * sizeof(float), stream);

  // xn = LN(x; gnn)  (f16 only)
  ln_kernel<<<NN / 8, 256, 0, stream>>>(x, gnn_g, gnn_b, nullptr, xn16);
  // message MLP + scatter-add
  msg_kernel<<<EE / 64, 128, 0, stream>>>(xn16, ei, ea, W1z, msg_b1, W2z, msg_b2, aggr);
  // update MLP
  upd_kernel<<<NN / 64, 128, 0, stream>>>(xn16, aggr, U1z, upd_b1, U2z, upd_b2, hloc);
  // attention
  qkv_kernel<<<NN / 64, 128, 0, stream>>>(x16, INz, in_b, q16, k16, v16b);
  attn_kernel<<<GG * HH, 128, 0, stream>>>(q16, k16, v16b, ao16);
  outproj_kernel<<<NN / 64, 128, 0, stream>>>(ao16, OWz, out_b, x, hloc, t32);
  // LN1 -> h1 (f32 + f16)
  ln_kernel<<<NN / 8, 256, 0, stream>>>(t32, n1_g, n1_b, h132, h116);
  // FFN + residual into t32
  ffn_kernel<<<NN / 64, 128, 0, stream>>>(h116, h132, F1z, ffn_b1, F2z, ffn_b2, t32);
  // LN2 -> d_out
  ln_kernel<<<NN / 8, 256, 0, stream>>>(t32, n2_g, n2_b, (float*)d_out, nullptr);
}
